// TreeBranch_56066503082477
// MI455X (gfx1250) — matrix operand order repaired and checked by reading, not compiled, not run on hardware
//
#include <hip/hip_runtime.h>
#include <hip/hip_bf16.h>
#include <stdint.h>

// ---------------------------------------------------------------------------
// TreeBranch on MI455X (gfx1250):
//  Pass 1: decisions -> leaf index -> bucket rows (atomic append, order-free)
//  Pass 2: weight split/transpose fp32 -> bf16 hi/lo, layout [leaf][d][f]
//  Pass 3: grouped GEMM per bucket with v_wmma_f32_16x16x32_bf16,
//          bf16x2 emulated-fp32 (a_hi*b_hi + a_hi*b_lo + a_lo*b_hi)
// Only the selected leaf is computed per row: 34 GFLOP instead of 137 GFLOP.
// Block tile 128M x 128N, 8 waves along M: one A-fragment conversion feeds
// 24 WMMAs; B fragments are contiguous b128 loads with immediate offsets.
// ---------------------------------------------------------------------------

typedef __attribute__((ext_vector_type(16))) __bf16 v16bf;
typedef __attribute__((ext_vector_type(8)))  float  v8f;
typedef __attribute__((ext_vector_type(4)))  float  v4f;

#define N_SAMPLES 32768
#define N_FEAT    1024
#define N_OUT     512
#define MT_BLOCK  128     // rows per block (8 waves x 16)
#define NT_BLOCK  128     // cols per block (8 tiles of 16 per wave)

// ---- Pass 2: leaf_w[k][f][d] (f32) -> bt_hi/bt_lo[k][d][f] (bf16) ----------
__global__ void prep_weights(const float* __restrict__ leaf_w,
                             __bf16* __restrict__ bt_hi,
                             __bf16* __restrict__ bt_lo) {
  int tid = blockIdx.x * 256 + threadIdx.x;   // 4*512*1024 = 2M threads
  int f = tid & (N_FEAT - 1);
  int r = tid >> 10;
  int d = r & (N_OUT - 1);
  int k = r >> 9;
  float w = leaf_w[((size_t)(k * N_FEAT + f)) * N_OUT + d];
  __bf16 hi = (__bf16)w;
  __bf16 lo = (__bf16)(w - (float)hi);
  size_t o = ((size_t)(k * N_OUT + d)) * N_FEAT + f;   // f-contiguous writes
  bt_hi[o] = hi;
  bt_lo[o] = lo;
}

// ---- Pass 1: decisions + bucketing ----------------------------------------
__global__ void decide(const float* __restrict__ xs,
                       const float* __restrict__ dec_w,
                       const float* __restrict__ dec_b,
                       int* __restrict__ cnt,
                       int* __restrict__ perm) {
  int lane = threadIdx.x & 31;
  int wave = threadIdx.x >> 5;
  int row = blockIdx.x * 8 + wave;            // one row per wave32
  const float* x = xs + (size_t)row * N_FEAT;
  float s0 = 0.f, s1 = 0.f, s2 = 0.f;
#pragma unroll 4
  for (int i = 0; i < N_FEAT / 32; ++i) {
    int f = i * 32 + lane;
    float xv = x[f];
    s0 += xv * dec_w[f];
    s1 += xv * dec_w[N_FEAT + f];
    s2 += xv * dec_w[2 * N_FEAT + f];
  }
#pragma unroll
  for (int off = 16; off; off >>= 1) {
    s0 += __shfl_xor(s0, off, 32);
    s1 += __shfl_xor(s1, off, 32);
    s2 += __shfl_xor(s2, off, 32);
  }
  if (lane == 0) {
    bool d0 = (s0 + dec_b[0]) > 0.f;
    bool d1 = (s1 + dec_b[1]) > 0.f;
    bool d2 = (s2 + dec_b[2]) > 0.f;
    int leaf = d0 ? (d2 ? 3 : 2) : (d1 ? 1 : 0);
    int pos = atomicAdd(&cnt[leaf], 1);
    perm[leaf * N_SAMPLES + pos] = row;
  }
}

// ---- Pass 3: grouped GEMM over buckets ------------------------------------
__global__ __launch_bounds__(256)
void leaf_gemm(const float* __restrict__ xs,
               const __bf16* __restrict__ bt_hi,
               const __bf16* __restrict__ bt_lo,
               const float* __restrict__ leaf_b,
               const int* __restrict__ cnt,
               const int* __restrict__ perm,
               float* __restrict__ out) {
  // Map blockIdx.y -> (leaf, mtile) by walking bucket tile counts.
  int cc[4] = {cnt[0], cnt[1], cnt[2], cnt[3]};
  int slot = blockIdx.y;
  int leaf = -1, mtile = 0, cn = 0;
#pragma unroll
  for (int k = 0; k < 4; ++k) {
    int tk = (cc[k] + MT_BLOCK - 1) / MT_BLOCK;
    if (leaf < 0) {
      if (slot < tk) { leaf = k; mtile = slot; cn = cc[k]; }
      else slot -= tk;
    }
  }
  if (leaf < 0) return;

  int lane = threadIdx.x & 31;
  int wave = threadIdx.x >> 5;
  int m0 = mtile * MT_BLOCK + wave * 16;         // row offset in bucket
  int n0 = blockIdx.x * NT_BLOCK;                // output column base

  const int* pbase = perm + leaf * N_SAMPLES;
  int gi = m0 + (lane & 15);
  int row = pbase[(gi < cn) ? gi : 0];           // fallback to a valid row
  // A fragment (16-bit A 16x32): lanes 0-15 hold K 0-15 of row=lane,
  // lanes 16-31 hold K 16-31 of row=lane-16.
  const float* arow = xs + (size_t)row * N_FEAT + ((lane & 16) ? 16 : 0);
  // B fragment (B 32x16): lanes 0-15 = col, K 0-15; lanes 16-31 = col, K 16-31.
  size_t colbase = ((size_t)(leaf * N_OUT + n0 + (lane & 15))) * N_FEAT
                 + ((lane & 16) ? 16 : 0);
  const __bf16* bh = bt_hi + colbase;
  const __bf16* bl = bt_lo + colbase;

  v8f acc[8] = {};

  for (int k0 = 0; k0 < N_FEAT; k0 += 32) {
    __builtin_prefetch(arow + k0 + 32, 0, 1);

    v4f a0 = *(const v4f*)(arow + k0);           // K +0..3
    v4f a1 = *(const v4f*)(arow + k0 + 4);       // K +4..7
    v4f a2 = *(const v4f*)(arow + k0 + 8);       // K +8..11
    v4f a3 = *(const v4f*)(arow + k0 + 12);      // K +12..15
    v16bf ahi, alo;
#pragma unroll
    for (int j = 0; j < 4; ++j) {
      float f0 = a0[j]; __bf16 h0 = (__bf16)f0;
      ahi[j]      = h0; alo[j]      = (__bf16)(f0 - (float)h0);
      float f1 = a1[j]; __bf16 h1 = (__bf16)f1;
      ahi[4 + j]  = h1; alo[4 + j]  = (__bf16)(f1 - (float)h1);
      float f2 = a2[j]; __bf16 h2 = (__bf16)f2;
      ahi[8 + j]  = h2; alo[8 + j]  = (__bf16)(f2 - (float)h2);
      float f3 = a3[j]; __bf16 h3 = (__bf16)f3;
      ahi[12 + j] = h3; alo[12 + j] = (__bf16)(f3 - (float)h3);
    }

#pragma unroll
    for (int t = 0; t < 8; ++t) {
      v16bf bhi = *(const v16bf*)(bh + k0 + t * 16 * N_FEAT);  // imm offsets
      v16bf blo = *(const v16bf*)(bl + k0 + t * 16 * N_FEAT);
      acc[t] = __builtin_amdgcn_wmma_f32_16x16x32_bf16(
          false, ahi, false, bhi, (short)0, acc[t], false, false);
      acc[t] = __builtin_amdgcn_wmma_f32_16x16x32_bf16(
          false, alo, false, bhi, (short)0, acc[t], false, false);
      acc[t] = __builtin_amdgcn_wmma_f32_16x16x32_bf16(
          false, ahi, false, blo, (short)0, acc[t], false, false);
    }
  }

  // C layout: VGPR v holds M = v (lanes 0-15) / M = v+8 (lanes 16-31), N = lane%16.
  int nlane = lane & 15;
  int mbase = (lane & 16) ? 8 : 0;
#pragma unroll
  for (int t = 0; t < 8; ++t) {
    int n = n0 + t * 16 + nlane;
    float bias = leaf_b[leaf * N_OUT + n];
#pragma unroll
    for (int v = 0; v < 8; ++v) {
      int g = m0 + mbase + v;
      if (g < cn) {
        int orow = pbase[g];
        out[(size_t)orow * N_OUT + n] = acc[t][v] + bias;
      }
    }
  }
}

// ---------------------------------------------------------------------------
extern "C" void kernel_launch(void* const* d_in, const int* in_sizes, int n_in,
                              void* d_out, int out_size, void* d_ws, size_t ws_size,
                              hipStream_t stream) {
  const float* xs     = (const float*)d_in[0];
  const float* dec_w  = (const float*)d_in[1];
  const float* dec_b  = (const float*)d_in[2];
  const float* leaf_w = (const float*)d_in[3];
  const float* leaf_b = (const float*)d_in[4];
  float* out = (float*)d_out;

  char* ws = (char*)d_ws;
  int*   cnt   = (int*)ws;                                  // 16 B
  int*   perm  = (int*)(ws + 256);                          // 4*32768*4 = 512 KB
  __bf16* bt_hi = (__bf16*)(ws + (1u << 20));               // 4 MB
  __bf16* bt_lo = (__bf16*)(ws + (1u << 20) + (4u << 20));  // 4 MB

  hipMemsetAsync(cnt, 0, 4 * sizeof(int), stream);

  prep_weights<<<(4 * N_OUT * N_FEAT) / 256, 256, 0, stream>>>(leaf_w, bt_hi, bt_lo);
  decide<<<N_SAMPLES / 8, 256, 0, stream>>>(xs, dec_w, dec_b, cnt, perm);

  dim3 grid(N_OUT / NT_BLOCK, N_SAMPLES / MT_BLOCK + 4);    // 4 x 260
  leaf_gemm<<<grid, 256, 0, stream>>>(xs, bt_hi, bt_lo, leaf_b, cnt, perm, out);
}
